// ResModel_45707041964063
// MI455X (gfx1250) — matrix operand order repaired and checked by reading, not compiled, not run on hardware
//
#include <hip/hip_runtime.h>

// ---------------- problem constants (from reference) ----------------
#define N_OP    50000
#define N_CFG   5000
#define N_EDGE  100000
#define NG      4
#define NCONF   32            // configs per config-node
#define HDIM    32
#define OPFEAT  140
#define CFGFEAT 18
#define OPE     32
#define NROWS   (N_OP * NCONF)   // 1,600,000 (op,config) rows
#define GIN     50               // CFGFEAT + HDIM
#define NFK     192              // node-feature K (172 padded to 6*32)
#define NTILES  3125             // N_OP / 16

typedef __bf16 bf16_t;
typedef __attribute__((ext_vector_type(16))) __bf16 v16bf;
typedef __attribute__((ext_vector_type(8)))  __bf16 v8bf;
typedef __attribute__((ext_vector_type(8)))  float  v8f;
typedef __attribute__((ext_vector_type(2)))  float  f2;

__device__ __forceinline__ float leaky(float v) { return v > 0.f ? v : 0.2f * v; }

// WMMA 16-bit operand layouts (ISA 7.12.2), wave32:
//  A 16x32: lane&15 = M; per lane two contiguous K runs: [8h,8h+8) and [16+8h,24+8h)
//           (h = lane>>4) -> loadable as 2x16B chunks of a 64B bf16 row.
//  B 32x16: lane&15 = N; element e -> K = (e&7) + 8h + 16(e>>3), same runs as A (pre-swizzled below)
//  C f32 16x16: vgpr r, lane l -> M = r + 8h, N = l&15

// ========================= utility kernels =========================
__global__ void k_fill(float* p, float v, int n) {
  int t = blockIdx.x * blockDim.x + threadIdx.x;
  if (t < n) p[t] = v;
}

__global__ void k_rsqrt_inplace(float* p, int n) {
  int t = blockIdx.x * blockDim.x + threadIdx.x;
  if (t < n) p[t] = rsqrtf(p[t]);
}

// f32 [K,32] row-major weight -> WMMA-B-swizzled bf16 [nkt*2 tiles][32 lanes][16]
// tile tt = kstep*2 + ntile; element (lane,e) holds W[kstep*32 + (e&7) + 8*(lane>>4) + 16*(e>>3),
//                                                   ntile*16 + (lane&15)]
__global__ void k_swz_b(const float* __restrict__ src, bf16_t* __restrict__ dst,
                        int K, int nkt) {
  int t = blockIdx.x * blockDim.x + threadIdx.x;
  if (t >= nkt * 2 * 512) return;
  int e = t & 15, lane = (t >> 4) & 31, tt = t >> 9;
  int kstep = tt >> 1, ntile = tt & 1;
  int k = kstep * 32 + (e & 7) + 8 * (lane >> 4) + 16 * (e >> 3);
  int n = ntile * 16 + (lane & 15);
  dst[t] = (bf16_t)((k < K) ? src[k * 32 + n] : 0.f);
}

// cfg_on_op[config_dst[j]] += 100 * config_feats[j]   (dense [N_OP,32,18] f32)
__global__ void k_scatter_cfg(const float* __restrict__ cfg_feats,
                              const int* __restrict__ cfg_dst,
                              float* __restrict__ cfg_on_op) {
  int j = blockIdx.x;
  size_t db = (size_t)cfg_dst[j] * (NCONF * CFGFEAT);
  size_t sb = (size_t)j * (NCONF * CFGFEAT);
  for (int t = threadIdx.x; t < NCONF * CFGFEAT; t += blockDim.x)
    atomicAdd(&cfg_on_op[db + t], 100.f * cfg_feats[sb + t]);
}

// f32 [row][18] -> bf16 [row][32] zero-padded (prenet A operand)
__global__ void k_cvt_cfg(const float* __restrict__ cfg_on_op,
                          bf16_t* __restrict__ cfg_bf) {
  int t = blockIdx.x * blockDim.x + threadIdx.x;   // row*32 + k  (51.2M < 2^31)
  int row = t >> 5, k = t & 31;
  cfg_bf[t] = (bf16_t)((k < CFGFEAT) ? cfg_on_op[(size_t)row * CFGFEAT + k] : 0.f);
}

// node feature matrix bf16 [N_OP,192]: [op_feats(140) | emb[op_codes](32) | 0-pad]
__global__ void k_build_nf(const float* __restrict__ op_feats,
                           const float* __restrict__ emb,
                           const int* __restrict__ op_codes,
                           bf16_t* __restrict__ nf) {
  int i = blockIdx.x;
  int k = threadIdx.x;           // 192 threads
  float v = 0.f;
  if (k < OPFEAT)            v = op_feats[(size_t)i * OPFEAT + k];
  else if (k < OPFEAT + OPE) v = emb[(size_t)op_codes[i] * OPE + (k - OPFEAT)];
  nf[(size_t)i * NFK + k] = (bf16_t)v;
}

// deg = 1 + in-degree + out-degree (buffer pre-filled with 1.0)
__global__ void k_deg(const int* __restrict__ src, const int* __restrict__ dst,
                      float* deg) {
  int e = blockIdx.x * blockDim.x + threadIdx.x;
  if (e < N_EDGE) {
    atomicAdd(&deg[src[e]], 1.f);
    atomicAdd(&deg[dst[e]], 1.f);
  }
}

// ========== nodecontrib GEMM (WMMA): [N_OP,172]@W1[18:,:] + b1 ==========
// Per wave: 16 op rows, K = 192 (6 steps x 2 N-tiles = 12 WMMAs).
__global__ __launch_bounds__(256)
void k_nodemm(const bf16_t* __restrict__ nf,
              const bf16_t* __restrict__ wswz,    // 12 tiles
              const float* __restrict__ b1,
              float* __restrict__ nodecontrib) {
  int lane = threadIdx.x & 31, wave = threadIdx.x >> 5;
  int tile = blockIdx.x * 8 + wave;
  if (tile >= NTILES) return;                     // wave-uniform guard
  int i0 = tile * 16;
  int half = lane >> 4, m = lane & 15, n = m;

  const v8bf* rowv = (const v8bf*)(nf + (size_t)(i0 + m) * NFK);  // 384B rows, 16B aligned
  const v16bf* wv = (const v16bf*)wswz;
  v8f c0 = {}, c1 = {};
#pragma unroll
  for (int ks = 0; ks < 6; ++ks) {
    v8bf alo = rowv[ks * 4 + half];
    v8bf ahi = rowv[ks * 4 + 2 + half];
    v16bf a;
#pragma unroll
    for (int e = 0; e < 8; ++e) { a[e] = alo[e]; a[8 + e] = ahi[e]; }
    v16bf bt0 = wv[(ks * 2 + 0) * 32 + lane];
    v16bf bt1 = wv[(ks * 2 + 1) * 32 + lane];
    c0 = __builtin_amdgcn_wmma_f32_16x16x32_bf16(false, a, false, bt0, (short)0, c0, false, false);
    c1 = __builtin_amdgcn_wmma_f32_16x16x32_bf16(false, a, false, bt1, (short)0, c1, false, false);
  }
  float bb0 = b1[n], bb1 = b1[n + 16];
#pragma unroll
  for (int r = 0; r < 8; ++r) {
    size_t o = (size_t)(i0 + r + 8 * half) * 32;
    nodecontrib[o + n]      = c0[r] + bb0;
    nodecontrib[o + n + 16] = c1[r] + bb1;
  }
}

// ===================== fused prenet (WMMA) ========================
// Per wave: 16 (op,cfg) rows (never straddles an op).  A tile (16x32 bf16, 1KB,
// contiguous) is staged LDS-ward with global_load_async_to_lds_b128 (ASYNCcnt),
// B tiles are single swizzled v16bf loads.  Stage1 +nodecontrib(b1) -> act ->
// LDS relayout -> Stage2 W2 +b2 -> act -> x.   Grid exact: 1.6M/16/8 = 12500.
__global__ __launch_bounds__(256)
void k_prenet(const bf16_t* __restrict__ cfg_bf,
              const float* __restrict__ nodecontrib,
              const bf16_t* __restrict__ w1swz,   // 2 tiles
              const bf16_t* __restrict__ w2swz,   // 2 tiles
              const float* __restrict__ b2v,
              float* __restrict__ x) {
  __shared__ __attribute__((aligned(64))) bf16_t lds_a[8][16 * 32];
  __shared__ __attribute__((aligned(64))) bf16_t lds_h[8][16 * 32];
  int lane = threadIdx.x & 31, wave = threadIdx.x >> 5;
  int r0 = (blockIdx.x * 8 + wave) * 16;
  int half = lane >> 4, m = lane & 15, n = m;
  int iop = r0 >> 5;                       // wave-uniform op index

  // ---- async copy of the wave's A tile into its private LDS slab ----
  unsigned ldsa = (unsigned)(size_t)&lds_a[wave][0] + (unsigned)(lane * 16);
  unsigned long long ga =
      (unsigned long long)(size_t)(cfg_bf + (size_t)r0 * 32) + (unsigned long long)(lane * 16);
  asm volatile("global_load_async_to_lds_b128 %0, %1, off" :: "v"(ldsa), "v"(ga) : "memory");
  asm volatile("global_load_async_to_lds_b128 %0, %1, off offset:512" :: "v"(ldsa), "v"(ga) : "memory");

  // overlap: B tiles + epilogue operands while the async copy is in flight
  const v16bf* w1v = (const v16bf*)w1swz;
  v16bf bt0 = w1v[lane];
  v16bf bt1 = w1v[32 + lane];
  float nc0 = nodecontrib[(size_t)iop * 32 + n];
  float nc1 = nodecontrib[(size_t)iop * 32 + 16 + n];
  const v16bf* w2v = (const v16bf*)w2swz;
  v16bf b20 = w2v[lane];
  v16bf b21 = w2v[32 + lane];
  float bb0 = b2v[n], bb1 = b2v[n + 16];

  asm volatile("s_wait_asynccnt 0" ::: "memory");
  const v8bf* arow = (const v8bf*)&lds_a[wave][m * 32];
  v8bf alo = arow[half], ahi = arow[2 + half];
  v16bf a;
#pragma unroll
  for (int e = 0; e < 8; ++e) { a[e] = alo[e]; a[8 + e] = ahi[e]; }

  v8f c0 = {}, c1 = {};
  c0 = __builtin_amdgcn_wmma_f32_16x16x32_bf16(false, a, false, bt0, (short)0, c0, false, false);
  c1 = __builtin_amdgcn_wmma_f32_16x16x32_bf16(false, a, false, bt1, (short)0, c1, false, false);

  // ---- epilogue 1: + nodecontrib, leaky, relayout via LDS ----
  bf16_t* h = lds_h[wave];
#pragma unroll
  for (int r = 0; r < 8; ++r) {
    int mrow = r + 8 * half;
    h[mrow * 32 + n]      = (bf16_t)leaky(c0[r] + nc0);
    h[mrow * 32 + n + 16] = (bf16_t)leaky(c1[r] + nc1);
  }
  const v8bf* hrow = (const v8bf*)&lds_h[wave][m * 32];
  v8bf hlo = hrow[half], hhi = hrow[2 + half];
  v16bf a2;
#pragma unroll
  for (int e = 0; e < 8; ++e) { a2[e] = hlo[e]; a2[8 + e] = hhi[e]; }

  v8f d0 = {}, d1 = {};
  d0 = __builtin_amdgcn_wmma_f32_16x16x32_bf16(false, a2, false, b20, (short)0, d0, false, false);
  d1 = __builtin_amdgcn_wmma_f32_16x16x32_bf16(false, a2, false, b21, (short)0, d1, false, false);
#pragma unroll
  for (int r = 0; r < 8; ++r) {
    size_t g = (size_t)(r0 + r + 8 * half) * HDIM;
    x[g + n]      = leaky(d0[r] + bb0);
    x[g + n + 16] = leaky(d1[r] + bb1);
  }
}

// ============= graph-conv propagation (a_times_x) =================
// ybuf/yout[g,0:50] = concat(cfg_on_op[g], x[g]) * dinv[op(g)]
__global__ __launch_bounds__(512)
void k_scale_concat(const float* __restrict__ cfg_on_op,
                    const float* __restrict__ x,
                    const float* __restrict__ dinv,
                    float* __restrict__ ybuf, float* __restrict__ yout) {
  int row = blockIdx.x * 8 + (threadIdx.x >> 6);
  int d = threadIdx.x & 63;
  if (row >= NROWS || d >= GIN) return;
  float di = dinv[row >> 5];
  float v = ((d < CFGFEAT) ? cfg_on_op[(size_t)row * CFGFEAT + d]
                           : x[(size_t)row * HDIM + (d - CFGFEAT)]) * di;
  size_t idx = (size_t)row * GIN + d;
  ybuf[idx] = v;
  yout[idx] = v;
}

// one block per edge: scatter-add the 32x50 row block in both directions
__global__ __launch_bounds__(256)
void k_edge(const int* __restrict__ src, const int* __restrict__ dst,
            const float* __restrict__ ybuf, float* __restrict__ yout) {
  int e = blockIdx.x;
  size_t bs = (size_t)src[e] * (NCONF * GIN);
  size_t bt = (size_t)dst[e] * (NCONF * GIN);
  __builtin_prefetch(ybuf + bs + (size_t)threadIdx.x * 4, 0, 1);  // global_prefetch_b8
  __builtin_prefetch(ybuf + bt + (size_t)threadIdx.x * 4, 0, 1);
  for (int k = threadIdx.x; k < NCONF * GIN; k += 256) {
    atomicAdd(&yout[bt + k], ybuf[bs + k]);
    atomicAdd(&yout[bs + k], ybuf[bt + k]);
  }
}

// ================ fused graph-conv MLP (WMMA) =====================
// A = yout[g,0:50] * dinv (trailing D^-1/2 folded in), K padded 50->64 as two
// WMMA steps; rows are 8B-aligned so A loads are naturally-aligned float2s.
__global__ __launch_bounds__(256)
void k_gc(const float* __restrict__ yprop, const float* __restrict__ dinv,
          const bf16_t* __restrict__ gw1swz,   // 4 tiles (Kpad 64)
          const float* __restrict__ gb1,
          const bf16_t* __restrict__ gw2swz,   // 2 tiles
          const float* __restrict__ gb2,
          float* __restrict__ x) {
  __shared__ __attribute__((aligned(64))) bf16_t lds_h[8][16 * 32];
  int lane = threadIdx.x & 31, wave = threadIdx.x >> 5;
  int r0 = (blockIdx.x * 8 + wave) * 16;
  int half = lane >> 4, m = lane & 15, n = m;
  float di = dinv[r0 >> 5];                       // wave-uniform (scalar load)

  const f2* yrow2 = (const f2*)(yprop + (size_t)(r0 + m) * GIN);  // 200B rows, 8B aligned

  // ---- stage 1 A, K-step 0: runs K[8h,8h+8) and K[16+8h,24+8h), all < 50 ----
  v16bf a0;
#pragma unroll
  for (int j = 0; j < 4; ++j) {
    f2 p = yrow2[4 * half + j];
    a0[2 * j]     = (bf16_t)(p.x * di);
    a0[2 * j + 1] = (bf16_t)(p.y * di);
    f2 q = yrow2[8 + 4 * half + j];
    a0[8 + 2 * j]     = (bf16_t)(q.x * di);
    a0[8 + 2 * j + 1] = (bf16_t)(q.y * di);
  }
  // ---- K-step 1: run K[32+8h,..) valid; run K[48+8h,..): only K=48,49 ----
  v16bf a1;
#pragma unroll
  for (int j = 0; j < 4; ++j) {
    f2 p = yrow2[16 + 4 * half + j];
    a1[2 * j]     = (bf16_t)(p.x * di);
    a1[2 * j + 1] = (bf16_t)(p.y * di);
  }
  {
    f2 tail = yrow2[24];                          // K = 48,49 (always in-bounds)
    float t0 = (half == 0) ? tail.x * di : 0.f;
    float t1 = (half == 0) ? tail.y * di : 0.f;
    a1[8] = (bf16_t)t0;
    a1[9] = (bf16_t)t1;
#pragma unroll
    for (int e = 2; e < 8; ++e) a1[8 + e] = (bf16_t)0.f;
  }

  const v16bf* wv = (const v16bf*)gw1swz;
  v16bf bt00 = wv[lane], bt01 = wv[32 + lane];
  v16bf bt10 = wv[64 + lane], bt11 = wv[96 + lane];
  float gbb0 = gb1[n], gbb1 = gb1[n + 16];

  v8f c0 = {}, c1 = {};
  c0 = __builtin_amdgcn_wmma_f32_16x16x32_bf16(false, a0, false, bt00, (short)0, c0, false, false);
  c1 = __builtin_amdgcn_wmma_f32_16x16x32_bf16(false, a0, false, bt01, (short)0, c1, false, false);
  c0 = __builtin_amdgcn_wmma_f32_16x16x32_bf16(false, a1, false, bt10, (short)0, c0, false, false);
  c1 = __builtin_amdgcn_wmma_f32_16x16x32_bf16(false, a1, false, bt11, (short)0, c1, false, false);

  bf16_t* h = lds_h[wave];
#pragma unroll
  for (int r = 0; r < 8; ++r) {
    int mrow = r + 8 * half;
    h[mrow * 32 + n]      = (bf16_t)leaky(c0[r] + gbb0);
    h[mrow * 32 + n + 16] = (bf16_t)leaky(c1[r] + gbb1);
  }
  const v8bf* hrow = (const v8bf*)&lds_h[wave][m * 32];
  v8bf hlo = hrow[half], hhi = hrow[2 + half];
  v16bf a2;
#pragma unroll
  for (int e = 0; e < 8; ++e) { a2[e] = hlo[e]; a2[8 + e] = hhi[e]; }

  const v16bf* w2v = (const v16bf*)gw2swz;
  v16bf b20 = w2v[lane], b21 = w2v[32 + lane];
  float bb0 = gb2[n], bb1 = gb2[n + 16];

  v8f d0 = {}, d1 = {};
  d0 = __builtin_amdgcn_wmma_f32_16x16x32_bf16(false, a2, false, b20, (short)0, d0, false, false);
  d1 = __builtin_amdgcn_wmma_f32_16x16x32_bf16(false, a2, false, b21, (short)0, d1, false, false);
#pragma unroll
  for (int r = 0; r < 8; ++r) {
    size_t g = (size_t)(r0 + r + 8 * half) * HDIM;
    x[g + n]      += leaky(d0[r] + bb0);          // residual
    x[g + n + 16] += leaky(d1[r] + bb1);
  }
}

// ====================== graph pooling =============================
__global__ __launch_bounds__(1024)
void k_pool_op(const float* __restrict__ x, const int* __restrict__ gid,
               float* __restrict__ op_sum, float* __restrict__ cnt) {
  __shared__ int gsh[128];
  int t = threadIdx.x;
  int base = blockIdx.x * 128;
  int lim = min(128, N_OP - base);
  if (t < lim) gsh[t] = gid[base + t];
  __syncthreads();
  float acc[NG] = {0.f, 0.f, 0.f, 0.f};
  for (int o = 0; o < lim; ++o) {
    float v = x[(size_t)(base + o) * (NCONF * HDIM) + t];
    int gg = gsh[o];
    acc[0] += (gg == 0) ? v : 0.f;
    acc[1] += (gg == 1) ? v : 0.f;
    acc[2] += (gg == 2) ? v : 0.f;
    acc[3] += (gg == 3) ? v : 0.f;
  }
#pragma unroll
  for (int g = 0; g < NG; ++g) atomicAdd(&op_sum[g * (NCONF * HDIM) + t], acc[g]);
  if (t == 0) {
    int c[NG] = {0, 0, 0, 0};
    for (int o = 0; o < lim; ++o) c[gsh[o]]++;
#pragma unroll
    for (int g = 0; g < NG; ++g) if (c[g]) atomicAdd(&cnt[g], (float)c[g]);
  }
}

__global__ __launch_bounds__(1024)
void k_pool_cfg(const float* __restrict__ x, const int* __restrict__ cdst,
                const int* __restrict__ gidc, float* __restrict__ cfg_sum) {
  __shared__ int gsh[128];
  __shared__ int rsh[128];
  int t = threadIdx.x;
  int base = blockIdx.x * 128;
  int lim = min(128, N_CFG - base);
  if (t < lim) { gsh[t] = gidc[base + t]; rsh[t] = cdst[base + t]; }
  __syncthreads();
  float acc[NG] = {0.f, 0.f, 0.f, 0.f};
  for (int o = 0; o < lim; ++o) {
    float v = x[(size_t)rsh[o] * (NCONF * HDIM) + t];
    int gg = gsh[o];
    acc[0] += (gg == 0) ? v : 0.f;
    acc[1] += (gg == 1) ? v : 0.f;
    acc[2] += (gg == 2) ? v : 0.f;
    acc[3] += (gg == 3) ? v : 0.f;
  }
#pragma unroll
  for (int g = 0; g < NG; ++g) atomicAdd(&cfg_sum[g * (NCONF * HDIM) + t], acc[g]);
}

// postnet: pooled[4,32,96] @ post_w1[96,32] -> leaky -> @ post_w2[32] -> out[4,32]
__global__ void k_post(const float* __restrict__ op_sum,
                       const float* __restrict__ cfg_sum,
                       const float* __restrict__ cnt,
                       const float* __restrict__ post_w1,
                       const float* __restrict__ post_w2,
                       float* __restrict__ out) {
  int t = threadIdx.x;
  if (t >= NG * NCONF) return;
  int g = t >> 5, c = t & 31;
  const float* os = op_sum + g * (NCONF * HDIM) + c * HDIM;
  const float* cs = cfg_sum + g * (NCONF * HDIM) + c * HDIM;
  float inv_cnt = 1.f / fmaxf(cnt[g], 1.f);
  float ss_o = 0.f, ss_c = 0.f;
  for (int h = 0; h < HDIM; ++h) { ss_o += os[h] * os[h]; ss_c += cs[h] * cs[h]; }
  float ro = rsqrtf(fmaxf(ss_o, 1e-12f));
  float rc = rsqrtf(fmaxf(ss_c, 1e-12f));
  float pooled[3 * HDIM];
  for (int h = 0; h < HDIM; ++h) {
    pooled[h]            = os[h] * inv_cnt;
    pooled[HDIM + h]     = os[h] * ro;
    pooled[2 * HDIM + h] = cs[h] * rc;
  }
  float res = 0.f;
  for (int n = 0; n < HDIM; ++n) {
    float a = 0.f;
    for (int k = 0; k < 3 * HDIM; ++k) a += pooled[k] * post_w1[k * HDIM + n];
    res += leaky(a) * post_w2[n];
  }
  out[t] = res;
}

// ========================= launcher ===============================
extern "C" void kernel_launch(void* const* d_in, const int* in_sizes, int n_in,
                              void* d_out, int out_size, void* d_ws, size_t ws_size,
                              hipStream_t stream) {
  (void)in_sizes; (void)n_in; (void)out_size; (void)ws_size;
  const float* op_feats     = (const float*)d_in[0];
  const float* config_feats = (const float*)d_in[1];
  const float* emb          = (const float*)d_in[2];
  const float* prenet_w1    = (const float*)d_in[3];
  const float* prenet_b1    = (const float*)d_in[4];
  const float* prenet_w2    = (const float*)d_in[5];
  const float* prenet_b2    = (const float*)d_in[6];
  const float* gc_w1[2] = {(const float*)d_in[7],  (const float*)d_in[11]};
  const float* gc_b1[2] = {(const float*)d_in[8],  (const float*)d_in[12]};
  const float* gc_w2[2] = {(const float*)d_in[9],  (const float*)d_in[13]};
  const float* gc_b2[2] = {(const float*)d_in[10], (const float*)d_in[14]};
  const float* post_w1      = (const float*)d_in[15];
  const float* post_w2      = (const float*)d_in[16];
  const int* op_codes     = (const int*)d_in[17];
  const int* feed_src     = (const int*)d_in[18];
  const int* feed_dst     = (const int*)d_in[19];
  const int* config_dst   = (const int*)d_in[20];
  const int* graph_id_op  = (const int*)d_in[21];
  const int* graph_id_cfg = (const int*)d_in[22];
  float* out = (float*)d_out;

  // ---- carve workspace (total ~1.09 GB) ----
  char* wp = (char*)d_ws;
  auto carve = [&](size_t bytes) -> void* {
    void* p = (void*)wp;
    wp += (bytes + 255) & ~(size_t)255;
    return p;
  };
  const size_t CFG_ON_OP_B = (size_t)N_OP * NCONF * CFGFEAT * 4;   // 115.2 MB
  const size_t CFG_BF_B    = (size_t)NROWS * 32 * 2;               // 102.4 MB
  const size_t X_B         = (size_t)N_OP * NCONF * HDIM * 4;      // 204.8 MB
  const size_t Y_B         = (size_t)N_OP * NCONF * GIN * 4;       // 320.0 MB
  const size_t NF_B        = (size_t)N_OP * NFK * 2;               //  19.2 MB
  float*  cfg_on_op   = (float*)carve(CFG_ON_OP_B);
  bf16_t* cfg_bf      = (bf16_t*)carve(CFG_BF_B);
  float*  x           = (float*)carve(X_B);
  float*  ybuf        = (float*)carve(Y_B);
  float*  yout        = (float*)carve(Y_B);
  bf16_t* nf          = (bf16_t*)carve(NF_B);
  float*  nodecontrib = (float*)carve((size_t)N_OP * HDIM * 4);
  float*  dinv        = (float*)carve((size_t)N_OP * 4);
  float*  sums        = (float*)carve((2 * NG * NCONF * HDIM + NG) * 4);
  float* op_sum  = sums;
  float* cfg_sum = sums + NG * NCONF * HDIM;
  float* cnt     = sums + 2 * NG * NCONF * HDIM;
  bf16_t* w1cfg_swz = (bf16_t*)carve(2 * 512 * 2);
  bf16_t* w2pre_swz = (bf16_t*)carve(2 * 512 * 2);
  bf16_t* nodew_swz = (bf16_t*)carve(12 * 512 * 2);
  bf16_t* gw1_swz[2] = {(bf16_t*)carve(4 * 512 * 2), (bf16_t*)carve(4 * 512 * 2)};
  bf16_t* gw2_swz[2] = {(bf16_t*)carve(2 * 512 * 2), (bf16_t*)carve(2 * 512 * 2)};

  // ---- init / preprocessing ----
  hipMemsetAsync(cfg_on_op, 0, CFG_ON_OP_B, stream);
  hipMemsetAsync(sums, 0, (2 * NG * NCONF * HDIM + NG) * 4, stream);
  k_fill<<<(N_OP + 255) / 256, 256, 0, stream>>>(dinv, 1.f, N_OP);
  k_swz_b<<<4, 256, 0, stream>>>(prenet_w1, w1cfg_swz, CFGFEAT, 1);           // K rows 0..17
  k_swz_b<<<4, 256, 0, stream>>>(prenet_w2, w2pre_swz, 32, 1);
  k_swz_b<<<24, 256, 0, stream>>>(prenet_w1 + CFGFEAT * 32, nodew_swz,
                                  OPFEAT + OPE, 6);                           // K rows 18..189
  for (int l = 0; l < 2; ++l) {
    k_swz_b<<<8, 256, 0, stream>>>(gc_w1[l], gw1_swz[l], GIN, 2);
    k_swz_b<<<4, 256, 0, stream>>>(gc_w2[l], gw2_swz[l], 32, 1);
  }
  k_scatter_cfg<<<N_CFG, 192, 0, stream>>>(config_feats, config_dst, cfg_on_op);
  k_cvt_cfg<<<(NROWS * 32) / 256, 256, 0, stream>>>(cfg_on_op, cfg_bf);
  k_build_nf<<<N_OP, NFK, 0, stream>>>(op_feats, emb, op_codes, nf);
  k_deg<<<(N_EDGE + 255) / 256, 256, 0, stream>>>(feed_src, feed_dst, dinv);
  k_rsqrt_inplace<<<(N_OP + 255) / 256, 256, 0, stream>>>(dinv, N_OP);

  // ---- nodecontrib GEMM (WMMA, K=192) ----
  k_nodemm<<<(NTILES + 7) / 8, 256, 0, stream>>>(nf, nodew_swz, prenet_b1, nodecontrib);

  // ---- prenet (WMMA + async-LDS staging) ----
  k_prenet<<<NROWS / (16 * 8), 256, 0, stream>>>(
      cfg_bf, nodecontrib, w1cfg_swz, w2pre_swz, prenet_b2, x);

  // ---- 2 graph-conv layers ----
  for (int l = 0; l < 2; ++l) {
    k_scale_concat<<<NROWS / 8, 512, 0, stream>>>(cfg_on_op, x, dinv, ybuf, yout);
    k_edge<<<N_EDGE, 256, 0, stream>>>(feed_src, feed_dst, ybuf, yout);
    k_gc<<<NROWS / (16 * 8), 256, 0, stream>>>(
        yout, dinv, gw1_swz[l], gc_b1[l], gw2_swz[l], gc_b2[l], x);
  }

  // ---- pooling + postnet ----
  k_pool_op<<<(N_OP + 127) / 128, 1024, 0, stream>>>(x, graph_id_op, op_sum, cnt);
  k_pool_cfg<<<(N_CFG + 127) / 128, 1024, 0, stream>>>(x, config_dst, graph_id_cfg, cfg_sum);
  k_post<<<1, 128, 0, stream>>>(op_sum, cfg_sum, cnt, post_w1, post_w2, out);
}
